// QuantLinearWB_3917010173956
// MI455X (gfx1250) — matrix operand order repaired and checked by reading, not compiled, not run on hardware
//
#include <hip/hip_runtime.h>

typedef __attribute__((ext_vector_type(16))) _Float16 v16h;
typedef __attribute__((ext_vector_type(8)))  _Float16 v8h;
typedef __attribute__((ext_vector_type(4)))  _Float16 v4h;
typedef __attribute__((ext_vector_type(2)))  _Float16 h2;
typedef __attribute__((ext_vector_type(8)))  float    v8f;

#define OUT_F   12288
#define IN_F    4096
#define GRP     128
#define NGRP    32
#define MROWS   8
#define QROWB   (NGRP * (GRP / 2))   // 2048 bytes of qcodes per output row

union AFrag { v16h v; v8h h[2]; };
union BFrag { v16h v; h2  p[8]; };

__global__ __launch_bounds__(128)
void qlinear_wmma_kernel(const float* __restrict__ x,
                         const unsigned char* __restrict__ qcodes,
                         const float* __restrict__ scales,
                         const float* __restrict__ zeros,
                         const float* __restrict__ bias,
                         float* __restrict__ out)
{
    // fp16 staged activations: 8 rows x 4096 cols = 64 KB
    __shared__ __align__(16) _Float16 xs[MROWS * IN_F];

    const int tid = threadIdx.x;

    // ---- stage x (fp32 -> fp16) into LDS once per block ----
    const float4* x4 = (const float4*)x;
    for (int i = tid; i < (MROWS * IN_F) / 4; i += 128) {
        float4 f = x4[i];
        v4h hv = { (_Float16)f.x, (_Float16)f.y, (_Float16)f.z, (_Float16)f.w };
        *(v4h*)(xs + i * 4) = hv;            // ds_store_b64
    }
    __syncthreads();

    const int wave  = tid >> 5;
    const int lane  = tid & 31;
    const int col   = lane & 15;             // N within tile
    const int khalf = lane >> 4;             // 0: K 0..15 of chunk, 1: K 16..31
    const int n     = (blockIdx.x * 4 + wave) * 16 + col;
    const int mrow  = lane & 7;              // rows 8..15 alias 0..7 (D rows discarded)

    const unsigned char* qrow = qcodes + (size_t)n * QROWB;
    const _Float16*      xrow = xs + mrow * IN_F;

    h2 k1024; k1024[0] = k1024[1] = (_Float16)1024.0f;

    v8f acc = {};

    for (int g = 0; g < NGRP; ++g) {
        const float s = scales[n * NGRP + g];
        const float z = zeros [n * NGRP + g];
        h2 spk;  spk[0]  = spk[1]  = (_Float16)s;
        h2 nzpk; nzpk[0] = nzpk[1] = (_Float16)(-z * s);

        const unsigned char* qg = qrow + g * (GRP / 2);

        #pragma unroll
        for (int c = 0; c < 4; ++c) {
            const int k0 = g * GRP + c * 32;

            // ---- B fragment: 16 nibbles for K = k0 + khalf*16 .. +15 ----
            const uint2 qb = *(const uint2*)(qg + c * 16 + khalf * 8);
            BFrag b;
            #pragma unroll
            for (int wi = 0; wi < 2; ++wi) {
                const unsigned int w  = wi ? qb.y : qb.x;
                const unsigned int lo =  w       & 0x0F0F0F0Fu;  // K odd  (low nibbles)
                const unsigned int hi = (w >> 4) & 0x0F0F0F0Fu;  // K even (high nibbles)
                #pragma unroll
                for (int by = 0; by < 4; ++by) {
#if __has_builtin(__builtin_amdgcn_perm)
                    // D = { 0x00, lo_by, 0x00, hi_by } ; sel 12 == 0x00 constant
                    const unsigned int sel = 0x0C000C00u |
                                             (unsigned)by | ((unsigned)(4 + by) << 16);
                    unsigned int pk = __builtin_amdgcn_perm(lo, hi, sel) | 0x64006400u;
#else
                    const unsigned int he = (hi >> (8 * by)) & 0xFu;
                    const unsigned int le = (lo >> (8 * by)) & 0xFu;
                    unsigned int pk = 0x64006400u | (le << 16) | he;
#endif
                    h2 hv = __builtin_bit_cast(h2, pk);   // {1024+q_even, 1024+q_odd}
                    h2 qv = hv - k1024;                   // exact (Sterbenz)
                    b.p[wi * 4 + by] = qv * spk + nzpk;   // v_pk_fma_f16
                }
            }

            // ---- A fragment from LDS (contiguous K = k0 + khalf*16 .. +15, matches B) ----
            const int ka = k0 + khalf * 16;
            AFrag a;
            a.h[0] = *(const v8h*)(xrow + ka);            // V0..V3
            a.h[1] = *(const v8h*)(xrow + ka + 8);        // V4..V7

            acc = __builtin_amdgcn_wmma_f32_16x16x32_f16(
                false, a.v, false, b.v, (short)0, acc, false, false);
        }
    }

    // lanes 0..15 hold D rows M = vgpr index (0..7), N = col
    if (lane < 16) {
        const float bb = bias[n];
        #pragma unroll
        for (int r = 0; r < MROWS; ++r)
            out[(size_t)r * OUT_F + n] = acc[r] + bb;
    }
}

extern "C" void kernel_launch(void* const* d_in, const int* in_sizes, int n_in,
                              void* d_out, int out_size, void* d_ws, size_t ws_size,
                              hipStream_t stream)
{
    const float*         x      = (const float*)d_in[0];
    const unsigned char* qcodes = (const unsigned char*)d_in[1];
    const float*         scales = (const float*)d_in[2];
    const float*         zeros  = (const float*)d_in[3];
    const float*         bias   = (const float*)d_in[4];
    float*               out    = (float*)d_out;

    dim3 grid(OUT_F / (4 * 16));   // 192 blocks; 4 waves/block; 16 cols per wave
    dim3 block(128);
    qlinear_wmma_kernel<<<grid, block, 0, stream>>>(x, qcodes, scales, zeros, bias, out);
}
